// GroupedQueryAttention_38379827757081
// MI455X (gfx1250) — matrix operand order repaired and checked by reading, not compiled, not run on hardware
//
#include <hip/hip_runtime.h>
#include <hip/hip_bf16.h>

typedef __attribute__((ext_vector_type(16))) _Float16 v16h;
typedef __attribute__((ext_vector_type(8)))  _Float16 v8h;
typedef __attribute__((ext_vector_type(8)))  float    v8f;
typedef __attribute__((ext_vector_type(4)))  unsigned int u32x4;
typedef __attribute__((ext_vector_type(8)))  int      i32x8;
typedef __attribute__((ext_vector_type(4)))  int      i32x4;

#define Bc   2
#define Sc   2048
#define Dc   2048
#define NQc  32
#define NKVc 8
#define HDc  64
#define DKVc 512   // NKV*HD

#if defined(__HIP_DEVICE_COMPILE__) && __has_builtin(__builtin_amdgcn_tensor_load_to_lds)
#define HAVE_TDM 1
#else
#define HAVE_TDM 0
#endif

// LDS B-tile row stride: 32 payload halves + 8 pad halves (TDM pad: 16DW interval, 4DW pad)
#define BT_STRIDE 40

// ---------------------------------------------------------------------------
// f32 -> f16 conversion (grid-stride)
// ---------------------------------------------------------------------------
__global__ void cvt_f32_f16(const float* __restrict__ in, _Float16* __restrict__ out, int n) {
    for (int i = blockIdx.x * blockDim.x + threadIdx.x; i < n; i += gridDim.x * blockDim.x)
        out[i] = (_Float16)in[i];
}

#if HAVE_TDM
// Issue one TDM 2D tile load: 32 (k, contiguous) x 128 (n rows) f16 elements,
// row stride K elements, into LDS at lds_off with row padding 4 DWORDs.
// D# layout per CDNA5 ISA 8.3/8.4. Wave-level op; EXEC ignored.
__device__ __forceinline__ void tdm_issue_b_tile(const _Float16* gsrc, unsigned lds_off, int K) {
    u32x4 g0;
    i32x8 g1;
    i32x4 g2 = {0, 0, 0, 0};
    i32x4 g3 = {0, 0, 0, 0};
    unsigned long long ga = (unsigned long long)(size_t)gsrc;
    g0[0] = 1u;                                              // count=1, user mode
    g0[1] = lds_off;                                         // LDS byte address
    g0[2] = (unsigned)ga;                                    // global_addr[31:0]
    g0[3] = (unsigned)((ga >> 32) & 0x01FFFFFFu) | (2u << 30); // addr[56:32] | type=2
    const unsigned dim0 = 1u << 30, dim1 = 1u << 30;         // huge dims: no OOB clip
    // data_size=2B (code 1), pad_enable, pad_interval=16DW (code 3), pad_amount=4DW (code 3)
    g1[0] = (int)((1u << 16) | (1u << 20) | (3u << 22) | (3u << 25));
    g1[1] = (int)((dim0 & 0xFFFFu) << 16);                   // tensor_dim0[15:0]
    g1[2] = (int)(((dim0 >> 16) & 0xFFFFu) | ((dim1 & 0xFFFFu) << 16));
    g1[3] = (int)(((dim1 >> 16) & 0xFFFFu) | (32u << 16));   // dim1 hi | tile_dim0=32
    g1[4] = 128;                                             // tile_dim1=128, tile_dim2=0
    g1[5] = K;                                               // tensor_dim0_stride[31:0]
    g1[6] = 0;
    g1[7] = 0;
#if __clang_major__ >= 23
    i32x8 g4 = {0, 0, 0, 0, 0, 0, 0, 0};
    __builtin_amdgcn_tensor_load_to_lds(g0, g1, g2, g3, g4, 0);
#else
    __builtin_amdgcn_tensor_load_to_lds(g0, g1, g2, g3, 0);
#endif
}
#endif

// ---------------------------------------------------------------------------
// C[b] (MxN) = A[b] (MxK, row major f16) * W^T  (W is NxK row major f16)
// Block: 8 waves, wave grid 2x4, 32x32 C tile per wave -> 64x128 block tile.
// B tile (128 rows x 32 k) staged into LDS via TDM (double buffered),
// A fragments loaded from global. 4 WMMAs per wave per k-step.
// ---------------------------------------------------------------------------
template <typename OutT>
__global__ void gemm_xwT(const _Float16* __restrict__ A, const _Float16* __restrict__ W,
                         OutT* __restrict__ C, int M, int N, int K) {
    __shared__ _Float16 bt[2][128 * BT_STRIDE];

    const int b = blockIdx.z;
    A += (size_t)b * M * K;
    C += (size_t)b * M * N;

    const int lane = threadIdx.x & 31;
    const int w    = threadIdx.x >> 5;        // 0..7
    const int wm   = w >> 2, wn = w & 3;      // 2x4 wave grid
    const int mBase = blockIdx.x * 64 + wm * 32;
    const int nBlk  = blockIdx.y * 128;

    const int half = lane >> 4;
    const int l16  = lane & 15;
    const int kA   = half * 16;               // A-frag k base within 32-chunk
    const int kB   = half * 16;               // B-frag k base within 32-chunk

    auto load_tile = [&](int buf, int k0) {
#if HAVE_TDM
        if (w == 0)
            tdm_issue_b_tile(W + (size_t)nBlk * K + k0,
                             (unsigned)(size_t)(void*)&bt[buf][0], K);
#else
        const int t = threadIdx.x;            // 256 threads -> 128 rows x 2 halves
        const int r = t >> 1, part = t & 1;
        const _Float16* src = W + (size_t)(nBlk + r) * K + k0 + part * 16;
        *(v8h*)&bt[buf][r * BT_STRIDE + part * 16] = *(const v8h*)src;
#endif
    };

    v8f acc[2][2] = {};
    load_tile(0, 0);

    for (int k0 = 0; k0 < K; k0 += 32) {
        const int cur = (k0 >> 5) & 1;
        const int nxt = cur ^ 1;
        if (k0 + 32 < K) load_tile(nxt, k0 + 32);
#if HAVE_TDM
        if (w == 0) {
            if (k0 + 32 < K) __builtin_amdgcn_s_wait_tensorcnt(1);
            else             __builtin_amdgcn_s_wait_tensorcnt(0);
        }
#endif
        __syncthreads();

        // A fragments: two 16-row sub-tiles
        v16h af[2];
#pragma unroll
        for (int mi = 0; mi < 2; ++mi) {
            const _Float16* ap = A + (size_t)(mBase + mi * 16 + l16) * K + k0 + kA;
            __builtin_prefetch(ap + 64, 0, 1);
            v8h alo = *(const v8h*)ap;
            v8h ahi = *(const v8h*)(ap + 8);
#pragma unroll
            for (int i = 0; i < 8; ++i) { af[mi][i] = alo[i]; af[mi][i + 8] = ahi[i]; }
        }
        // B fragments from LDS (16B-aligned: BT_STRIDE*2 = 80B rows, kB*2 in {0,32})
#pragma unroll
        for (int ni = 0; ni < 2; ++ni) {
            const int nl = wn * 32 + ni * 16 + l16;
            const _Float16* bp = &bt[cur][nl * BT_STRIDE + kB];
            v8h blo = *(const v8h*)bp;
            v8h bhi = *(const v8h*)(bp + 8);
            v16h bf;
#pragma unroll
            for (int i = 0; i < 8; ++i) { bf[i] = blo[i]; bf[i + 8] = bhi[i]; }
#pragma unroll
            for (int mi = 0; mi < 2; ++mi)
                acc[mi][ni] = __builtin_amdgcn_wmma_f32_16x16x32_f16(
                    false, af[mi], false, bf, (short)0, acc[mi][ni], false, false);
        }
        __syncthreads();
    }

#pragma unroll
    for (int mi = 0; mi < 2; ++mi)
#pragma unroll
        for (int ni = 0; ni < 2; ++ni)
#pragma unroll
            for (int r = 0; r < 8; ++r) {
                const int row = mBase + mi * 16 + r + half * 8;
                const int col = nBlk + wn * 32 + ni * 16 + l16;
                C[(size_t)row * N + col] = (OutT)acc[mi][ni][r];
            }
}

// ---------------------------------------------------------------------------
// In-place RoPE on (B, S, nh, HD) f16 buffer; one thread per (pair) element.
// ---------------------------------------------------------------------------
__global__ void rope_inplace(_Float16* __restrict__ buf, int nh, int total) {
    int i = blockIdx.x * blockDim.x + threadIdx.x;
    if (i >= total) return;
    int t  = i;
    int dd = t & 31;  t >>= 5;
    int hh = t % nh;  t /= nh;
    int s  = t % Sc;
    int bb = t / Sc;
    size_t base = (((size_t)bb * Sc + s) * nh + hh) * HDc;
    float x1 = (float)buf[base + dd];
    float x2 = (float)buf[base + dd + 32];
    // inv_freq = theta^(-dd/32),  ln(10000)/32 = 0.28782313662425572
    float ang = (float)s * __expf(-(float)dd * 0.28782313662425572f);
    float c = __cosf(ang), sn = __sinf(ang);
    buf[base + dd]      = (_Float16)(x1 * c - x2 * sn);
    buf[base + dd + 32] = (_Float16)(x2 * c + x1 * sn);
}

// ---------------------------------------------------------------------------
// V (b, s, kvh, hd) -> VT (b, kvh, hd, s): makes PV B-fragments contiguous.
// ---------------------------------------------------------------------------
__global__ void transpose_v(const _Float16* __restrict__ V, _Float16* __restrict__ VT, int n) {
    int i = blockIdx.x * blockDim.x + threadIdx.x;
    if (i >= n) return;
    int t   = i;
    int dd  = t & 63;   t >>= 6;
    int kvh = t & 7;    t >>= 3;
    int s   = t % Sc;
    int bb  = t / Sc;
    VT[(((size_t)(bb * NKVc + kvh) * HDc) + dd) * Sc + s] = V[i];
}

// ---------------------------------------------------------------------------
// Fused causal flash attention.
// Grid: (S/128, NQ, B). Block: 256 threads = 8 waves; each wave owns 16 q rows.
// Q: (b,s,NQ*HD) roped f16; K: (b,s,NKV*HD) roped f16; VT: (b,kvh,hd,s) f16.
// Output O: (b,s,NQ*HD) f16.
// ---------------------------------------------------------------------------
__global__ void flash_attn(const _Float16* __restrict__ Q, const _Float16* __restrict__ Kb,
                           const _Float16* __restrict__ VT, _Float16* __restrict__ Ob) {
    __shared__ _Float16 pbuf[8][16 * 36];     // per-wave 16x32 P tile, padded rows

    const int bb   = blockIdx.z;
    const int h    = blockIdx.y;
    const int kvh  = h >> 2;                  // GQA group of 4
    const int w    = threadIdx.x >> 5;
    const int lane = threadIdx.x & 31;
    const int half = lane >> 4;
    const int l16  = lane & 15;
    const int q0   = blockIdx.x * 128 + w * 16;
    const int kA   = half * 16;

    // ---- load Q fragments (16 rows x 64 d = 2 k-steps of 32), loop invariant
    v16h qa[2];
#pragma unroll
    for (int ks = 0; ks < 2; ++ks) {
        const _Float16* qp = Q + ((size_t)(bb * Sc + q0 + l16)) * Dc + h * HDc + ks * 32 + kA;
        v8h lo = *(const v8h*)qp;
        v8h hi = *(const v8h*)(qp + 8);
#pragma unroll
        for (int i = 0; i < 8; ++i) { qa[ks][i] = lo[i]; qa[ks][i + 8] = hi[i]; }
    }

    float mi[8], li[8];
    v8f acc[4] = {v8f{}, v8f{}, v8f{}, v8f{}};
#pragma unroll
    for (int r = 0; r < 8; ++r) { mi[r] = -3.0e38f; li[r] = 0.0f; }

    const int kvend = q0 + 16;                 // causal bound (exclusive)
    for (int kv0 = 0; kv0 < kvend; kv0 += 32) {
        // ---- scores: two 16x16 tiles over HD=64 (2 WMMA k-steps each)
        v8f s0 = {}, s1 = {};
#pragma unroll
        for (int t = 0; t < 2; ++t) {
            const int j = kv0 + t * 16 + l16;  // B-frag: lane column = kv index
            const _Float16* kp = Kb + ((size_t)(bb * Sc + j)) * DKVc + kvh * HDc + half * 16;
#pragma unroll
            for (int ks = 0; ks < 2; ++ks) {
                v8h lo = *(const v8h*)(kp + ks * 32);
                v8h hi = *(const v8h*)(kp + ks * 32 + 8);
                v16h bf;
#pragma unroll
                for (int i = 0; i < 8; ++i) { bf[i] = lo[i]; bf[i + 8] = hi[i]; }
                v8f& st = t ? s1 : s0;
                st = __builtin_amdgcn_wmma_f32_16x16x32_f16(false, qa[ks], false, bf,
                                                            (short)0, st, false, false);
            }
        }

        // ---- scale, causal mask, online softmax (rows live in 16-lane halves)
        float cand[8];
#pragma unroll
        for (int r = 0; r < 8; ++r) {
            const int irow = q0 + r + half * 8;
            const int j0 = kv0 + l16, j1 = j0 + 16;
            float v0 = s0[r] * 0.125f; if (j0 > irow) v0 = -3.0e38f;
            float v1 = s1[r] * 0.125f; if (j1 > irow) v1 = -3.0e38f;
            s0[r] = v0; s1[r] = v1;
            cand[r] = fmaxf(v0, v1);
        }
#pragma unroll
        for (int r = 0; r < 8; ++r)
#pragma unroll
            for (int m = 1; m < 16; m <<= 1)
                cand[r] = fmaxf(cand[r], __shfl_xor(cand[r], m, 32));

        float alpha[8], rsum[8];
#pragma unroll
        for (int r = 0; r < 8; ++r) {
            const float mn = fmaxf(mi[r], cand[r]);
            alpha[r] = __expf(mi[r] - mn);
            mi[r] = mn;
            const float p0 = __expf(s0[r] - mn);
            const float p1 = __expf(s1[r] - mn);
            s0[r] = p0; s1[r] = p1;
            rsum[r] = p0 + p1;
        }
#pragma unroll
        for (int r = 0; r < 8; ++r)
#pragma unroll
            for (int m = 1; m < 16; m <<= 1)
                rsum[r] += __shfl_xor(rsum[r], m, 32);
#pragma unroll
        for (int r = 0; r < 8; ++r) li[r] = li[r] * alpha[r] + rsum[r];
#pragma unroll
        for (int dt = 0; dt < 4; ++dt)
#pragma unroll
            for (int r = 0; r < 8; ++r) acc[dt][r] *= alpha[r];

        // ---- transpose P (C-layout -> A-layout) through per-wave LDS tile
        __builtin_amdgcn_wave_barrier();
#pragma unroll
        for (int r = 0; r < 8; ++r) {
            pbuf[w][(r + half * 8) * 36 + l16]      = (_Float16)s0[r];
            pbuf[w][(r + half * 8) * 36 + 16 + l16] = (_Float16)s1[r];
        }
        __builtin_amdgcn_wave_barrier();
        v16h pa;
#pragma unroll
        for (int e = 0; e < 8; ++e) {
            pa[e]     = pbuf[w][l16 * 36 + kA + e];
            pa[e + 8] = pbuf[w][l16 * 36 + kA + 8 + e];
        }
        __builtin_amdgcn_wave_barrier();

        // ---- PV: 4 output d-tiles, K=32 per WMMA; VT gives contiguous B-frags
#pragma unroll
        for (int dt = 0; dt < 4; ++dt) {
            const _Float16* vp = VT + (((size_t)(bb * NKVc + kvh) * HDc) + dt * 16 + l16) * Sc
                                    + kv0 + half * 16;
            v8h lo = *(const v8h*)vp;
            v8h hi = *(const v8h*)(vp + 8);
            v16h vb;
#pragma unroll
            for (int i = 0; i < 8; ++i) { vb[i] = lo[i]; vb[i + 8] = hi[i]; }
            acc[dt] = __builtin_amdgcn_wmma_f32_16x16x32_f16(false, pa, false, vb,
                                                             (short)0, acc[dt], false, false);
        }
    }

    // ---- epilogue: divide by row sums and store (b, s, h*HD + dd) f16
#pragma unroll
    for (int dt = 0; dt < 4; ++dt)
#pragma unroll
        for (int r = 0; r < 8; ++r) {
            const int row = q0 + r + half * 8;
            Ob[((size_t)(bb * Sc + row)) * Dc + h * HDc + dt * 16 + l16] =
                (_Float16)(acc[dt][r] / li[r]);
        }
}

// ---------------------------------------------------------------------------
extern "C" void kernel_launch(void* const* d_in, const int* in_sizes, int n_in,
                              void* d_out, int out_size, void* d_ws, size_t ws_size,
                              hipStream_t stream) {
    const float* x  = (const float*)d_in[0];
    const float* Wq = (const float*)d_in[2];
    const float* Wk = (const float*)d_in[3];
    const float* Wv = (const float*)d_in[4];
    const float* Wo = (const float*)d_in[5];
    float* out = (float*)d_out;

    char* ws = (char*)d_ws;
    size_t off = 0;
    auto take = [&](size_t elems) { _Float16* p = (_Float16*)(ws + off); off += elems * sizeof(_Float16); return p; };

    const size_t nX  = (size_t)Bc * Sc * Dc;          // 8.39M
    const size_t nWq = (size_t)Dc * Dc;               // 4.19M
    const size_t nWk = (size_t)DKVc * Dc;             // 1.05M
    const size_t nK  = (size_t)Bc * Sc * DKVc;        // 2.10M

    _Float16* xh   = take(nX);
    _Float16* wqh  = take(nWq);
    _Float16* wkh  = take(nWk);
    _Float16* wvh  = take(nWk);
    _Float16* woh  = take(nWq);
    _Float16* qh   = take(nX);
    _Float16* kh   = take(nK);
    _Float16* vh   = take(nK);
    _Float16* vth  = take(nK);
    _Float16* attn = take(nX);

    // 1) downconvert activations + weights to f16
    cvt_f32_f16<<<2048, 256, 0, stream>>>(x,  xh,  (int)nX);
    cvt_f32_f16<<<1024, 256, 0, stream>>>(Wq, wqh, (int)nWq);
    cvt_f32_f16<<<512,  256, 0, stream>>>(Wk, wkh, (int)nWk);
    cvt_f32_f16<<<512,  256, 0, stream>>>(Wv, wvh, (int)nWk);
    cvt_f32_f16<<<1024, 256, 0, stream>>>(Wo, woh, (int)nWq);

    // 2) QKV projections (WMMA GEMMs, TDM-staged B tiles)
    gemm_xwT<_Float16><<<dim3(Sc / 64, Dc / 128,   Bc), 256, 0, stream>>>(xh, wqh, qh, Sc, Dc,   Dc);
    gemm_xwT<_Float16><<<dim3(Sc / 64, DKVc / 128, Bc), 256, 0, stream>>>(xh, wkh, kh, Sc, DKVc, Dc);
    gemm_xwT<_Float16><<<dim3(Sc / 64, DKVc / 128, Bc), 256, 0, stream>>>(xh, wvh, vh, Sc, DKVc, Dc);

    // 3) RoPE on Q, K; transpose V for contiguous PV fragments
    rope_inplace<<<(Bc * Sc * NQc * 32) / 256,  256, 0, stream>>>(qh, NQc,  Bc * Sc * NQc * 32);
    rope_inplace<<<(Bc * Sc * NKVc * 32) / 256, 256, 0, stream>>>(kh, NKVc, Bc * Sc * NKVc * 32);
    transpose_v<<<(int)(nK / 256), 256, 0, stream>>>(vh, vth, (int)nK);

    // 4) fused causal flash attention (WMMA scores + WMMA PV)
    flash_attn<<<dim3(Sc / 128, NQc, Bc), 256, 0, stream>>>(qh, kh, vth, attn);

    // 5) output projection, fp32 result
    gemm_xwT<float><<<dim3(Sc / 64, Dc / 128, Bc), 256, 0, stream>>>(attn, woh, out, Sc, Dc, Dc);
}
